// Graph_Net_57277683860151
// MI455X (gfx1250) — matrix operand order repaired and checked by reading, not compiled, not run on hardware
//
#include <hip/hip_runtime.h>

typedef __attribute__((ext_vector_type(16))) _Float16 v16h;
typedef __attribute__((ext_vector_type(8)))  float    v8f;

#define N_NODES     100000
#define N_EDGES     1600000
#define HID         128
#define NUM_LAYERS  10
#define NUM_GRAPHS  64
#define NUM_CLASSES 2
#define NTILES      (N_NODES / 16)          // 6250 16-row tiles
#define WAVES_PER_BLOCK 8
#define BLOCK       (WAVES_PER_BLOCK * 32)  // 256 threads = 8 wave32

// ---------------------------------------------------------------------------
// zero fill
// ---------------------------------------------------------------------------
__global__ void zero_f32(float* __restrict__ p, int n) {
    int i = blockIdx.x * blockDim.x + threadIdx.x;
    if (i < n) p[i] = 0.0f;
}

// ---------------------------------------------------------------------------
// Edge gather + scatter-add: aggr[dst] += x[src]
// One wave per edge, one float4 per lane (32 lanes * 4 = 128 features).
// unsafeAtomicAdd -> global_atomic_add_f32 (no CAS loop); aggr stays in L2.
// ---------------------------------------------------------------------------
__global__ void gin_scatter(const float* __restrict__ x,
                            const int*   __restrict__ src,
                            const int*   __restrict__ dst,
                            float*       __restrict__ aggr) {
    long long g = (long long)blockIdx.x * blockDim.x + threadIdx.x;
    int edge = (int)(g >> 5);
    int lane = (int)(g & 31);
    if (edge >= N_EDGES) return;
    int s = src[edge];
    int d = dst[edge];
    const float4 v = *(const float4*)(x + (long long)s * HID + lane * 4);
    float* o = aggr + (long long)d * HID + lane * 4;
    unsafeAtomicAdd(o + 0, v.x);
    unsafeAtomicAdd(o + 1, v.y);
    unsafeAtomicAdd(o + 2, v.z);
    unsafeAtomicAdd(o + 3, v.w);
}

// ---------------------------------------------------------------------------
// Fused GIN MLP block for one layer:
//   h = (1+eps)*x + aggr
//   out = relu( relu(h@W1+b1) @ W2 + b2 [+ h if layer>0] ) + x
// One wave computes a 16x128 tile; K=128 -> 4 WMMA f16 k-blocks, N=128 -> 8
// n-tiles, two GEMM stages = 64 v_wmma_f32_16x16x32_f16 per tile.
// W1/W2 are repacked once per block into LDS in WMMA-B fragment order so the
// inner loop B load is a single contiguous 32B ds read per fragment.
// ---------------------------------------------------------------------------
__global__ void __launch_bounds__(BLOCK)
gin_mlp(const float* __restrict__ x,   const float* __restrict__ aggr,
        const float* __restrict__ eps_arr, int layer,
        const float* __restrict__ W1,  const float* __restrict__ b1,
        const float* __restrict__ W2,  const float* __restrict__ b2,
        int mlp_residual, float* __restrict__ xout)
{
    extern __shared__ _Float16 smem[];
    _Float16* w1p  = smem;              // 16384 halfs = 32 KB (B-frag order)
    _Float16* w2p  = smem + 16384;      // 16384 halfs = 32 KB
    _Float16* midb = smem + 32768;      // 8 waves * 2048 halfs = 32 KB

    const int tid = threadIdx.x;

    // ---- cooperative repack of W1/W2 (f32 row-major -> f16 B fragments) ----
    // Fragment e = (kb*8 + nt)*32 + lane, 16 halfs per (fragment, lane).
    // B layout (16-bit, 32x16): lane holds column n = nt*16 + (lane&15);
    // elems 0..7 = K kb*32+hi*8 .. +7, elems 8..15 = K kb*32+16+hi*8 .. +7,
    // where hi = lane>>4.
    for (int e = tid; e < 1024; e += BLOCK) {
        int kb  = e >> 8;
        int rem = e & 255;
        int nt  = rem >> 5;
        int pl  = rem & 31;
        int n   = nt * 16 + (pl & 15);
        int kbase = kb * 32 + (pl >> 4) * 8;
        _Float16* d1 = w1p + e * 16;
        _Float16* d2 = w2p + e * 16;
#pragma unroll
        for (int i = 0; i < 8; ++i) {
            d1[i]     = (_Float16)W1[(kbase + i) * HID + n];
            d1[8 + i] = (_Float16)W1[(kbase + 16 + i) * HID + n];
            d2[i]     = (_Float16)W2[(kbase + i) * HID + n];
            d2[8 + i] = (_Float16)W2[(kbase + 16 + i) * HID + n];
        }
    }
    __syncthreads();

    const int wave = tid >> 5;
    const int lane = tid & 31;
    const int tile = blockIdx.x * WAVES_PER_BLOCK + wave;
    if (tile >= NTILES) return;

    const float epsv = 1.0f + eps_arr[layer];
    const int r  = lane & 15;   // row within tile (A) / column (B,C,D)
    const int hi = lane >> 4;
    const long long rowbase = (long long)(tile * 16 + r) * HID;
    _Float16* mid = midb + wave * 2048;   // 16x128 f16, wave-private

    // ---- A1 fragments: h computed on the fly from global x/aggr ----
    v16h a1[4];
#pragma unroll
    for (int kb = 0; kb < 4; ++kb) {
        const int c0 = kb * 32 + hi * 8;
#pragma unroll
        for (int i = 0; i < 8; ++i) {
            float h0 = epsv * x[rowbase + c0 + i]      + aggr[rowbase + c0 + i];
            float h1 = epsv * x[rowbase + c0 + 16 + i] + aggr[rowbase + c0 + 16 + i];
            a1[kb][i]     = (_Float16)h0;
            a1[kb][8 + i] = (_Float16)h1;
        }
    }

    // ---- stage 1: mid = relu(h @ W1 + b1)  (written to LDS as f16) ----
#pragma unroll
    for (int nt = 0; nt < 8; ++nt) {
        v8f c = {};
#pragma unroll
        for (int kb = 0; kb < 4; ++kb) {
            v16h b = *(const v16h*)(w1p + ((kb * 8 + nt) * 32 + lane) * 16);
            c = __builtin_amdgcn_wmma_f32_16x16x32_f16(false, a1[kb], false, b,
                                                       (short)0, c, false, false);
        }
        const int n = nt * 16 + r;        // C/D: lane holds column n
        const float bias = b1[n];
#pragma unroll
        for (int v = 0; v < 8; ++v) {     // VGPR v -> row v + 8*hi
            float val = fmaxf(c[v] + bias, 0.0f);
            mid[(v + 8 * hi) * HID + n] = (_Float16)val;
        }
    }

    // ---- A2 fragments from the LDS transpose buffer ----
    v16h a2[4];
#pragma unroll
    for (int kb = 0; kb < 4; ++kb) {
        const int c0 = r * HID + kb * 32 + hi * 8;
#pragma unroll
        for (int i = 0; i < 8; ++i) {
            a2[kb][i]     = mid[c0 + i];
            a2[kb][8 + i] = mid[c0 + 16 + i];
        }
    }

    // ---- stage 2 + epilogue: out = relu(mid@W2 + b2 [+h]) + x ----
#pragma unroll
    for (int nt = 0; nt < 8; ++nt) {
        v8f c = {};
#pragma unroll
        for (int kb = 0; kb < 4; ++kb) {
            v16h b = *(const v16h*)(w2p + ((kb * 8 + nt) * 32 + lane) * 16);
            c = __builtin_amdgcn_wmma_f32_16x16x32_f16(false, a2[kb], false, b,
                                                       (short)0, c, false, false);
        }
        const int n = nt * 16 + r;
        const float bias = b2[n];
#pragma unroll
        for (int v = 0; v < 8; ++v) {
            const long long gi = (long long)(tile * 16 + v + 8 * hi) * HID + n;
            const float xv = x[gi];
            float val = c[v] + bias;
            if (mlp_residual) val += epsv * xv + aggr[gi];   // recompute h
            val = fmaxf(val, 0.0f);
            xout[gi] = val + xv;                             // block residual
        }
    }
}

// ---------------------------------------------------------------------------
// global mean pool: pooled[batch[n]] += x[n]; cnt[batch[n]] += 1
// ---------------------------------------------------------------------------
__global__ void pool_scatter(const float* __restrict__ x,
                             const int*   __restrict__ batch,
                             float* __restrict__ pooled,
                             float* __restrict__ cnt) {
    long long g = (long long)blockIdx.x * blockDim.x + threadIdx.x;
    if (g >= (long long)N_NODES * HID) return;
    int node = (int)(g >> 7);
    int f    = (int)(g & 127);
    int b    = batch[node];
    unsafeAtomicAdd(&pooled[b * HID + f], x[g]);
    if (f == 0) unsafeAtomicAdd(&cnt[b], 1.0f);
}

// ---------------------------------------------------------------------------
// classifier: logits = relu(pooled/cnt @ Wc1 + bc1) @ Wc2 + bc2
// one block per graph, 128 threads (tiny; VALU is fine here)
// ---------------------------------------------------------------------------
__global__ void classifier(const float* __restrict__ pooled,
                           const float* __restrict__ cnt,
                           const float* __restrict__ Wc1, const float* __restrict__ bc1,
                           const float* __restrict__ Wc2, const float* __restrict__ bc2,
                           float* __restrict__ out) {
    __shared__ float ph[HID];
    __shared__ float hd[HID];
    const int g = blockIdx.x;
    const int j = threadIdx.x;
    const float inv = 1.0f / fmaxf(cnt[g], 1.0f);
    ph[j] = pooled[g * HID + j] * inv;
    __syncthreads();
    float acc = bc1[j];
    for (int k = 0; k < HID; ++k) acc += ph[k] * Wc1[k * HID + j];
    hd[j] = fmaxf(acc, 0.0f);
    __syncthreads();
    if (j < NUM_CLASSES) {
        float acc2 = bc2[j];
        for (int k = 0; k < HID; ++k) acc2 += hd[k] * Wc2[k * NUM_CLASSES + j];
        out[g * NUM_CLASSES + j] = acc2;
    }
}

// ---------------------------------------------------------------------------
extern "C" void kernel_launch(void* const* d_in, const int* in_sizes, int n_in,
                              void* d_out, int out_size, void* d_ws, size_t ws_size,
                              hipStream_t stream) {
    const float* x0   = (const float*)d_in[0];
    const int*   edge = (const int*)  d_in[1];   // [2, N_EDGES]
    const int*   batch= (const int*)  d_in[2];
    const float* eps  = (const float*)d_in[3];
    const float* W1   = (const float*)d_in[4];
    const float* b1   = (const float*)d_in[5];
    const float* W2   = (const float*)d_in[6];
    const float* b2   = (const float*)d_in[7];
    const float* Wc1  = (const float*)d_in[8];
    const float* bc1  = (const float*)d_in[9];
    const float* Wc2  = (const float*)d_in[10];
    const float* bc2  = (const float*)d_in[11];
    const int* src = edge;
    const int* dst = edge + N_EDGES;

    const long long NF = (long long)N_NODES * HID;   // 12.8M floats
    float* ws     = (float*)d_ws;
    float* xA     = ws;
    float* xB     = ws + NF;
    float* aggr   = ws + 2 * NF;
    float* pooled = ws + 3 * NF;
    float* cnt    = pooled + NUM_GRAPHS * HID;

    const int zgrid = (int)((NF + 255) / 256);
    const int sgrid = (int)(((long long)N_EDGES * 32 + 255) / 256);
    const int mgrid = (NTILES + WAVES_PER_BLOCK - 1) / WAVES_PER_BLOCK;
    const size_t lds = (size_t)(16384 + 16384 + WAVES_PER_BLOCK * 2048) * sizeof(_Float16);

    const float* xcur = x0;
    float* xnext = xA;
    for (int i = 0; i < NUM_LAYERS; ++i) {
        zero_f32<<<zgrid, 256, 0, stream>>>(aggr, (int)NF);
        gin_scatter<<<sgrid, 256, 0, stream>>>(xcur, src, dst, aggr);
        gin_mlp<<<mgrid, BLOCK, lds, stream>>>(
            xcur, aggr, eps, i,
            W1 + (long long)i * HID * HID, b1 + i * HID,
            W2 + (long long)i * HID * HID, b2 + i * HID,
            (i > 0) ? 1 : 0, xnext);
        xcur  = xnext;
        xnext = (xnext == xA) ? xB : xA;
    }

    zero_f32<<<(NUM_GRAPHS * HID + NUM_GRAPHS + 255) / 256, 256, 0, stream>>>(
        pooled, NUM_GRAPHS * HID + NUM_GRAPHS);
    pool_scatter<<<zgrid, 256, 0, stream>>>(xcur, batch, pooled, cnt);
    classifier<<<NUM_GRAPHS, HID, 0, stream>>>(pooled, cnt, Wc1, bc1, Wc2, bc2,
                                               (float*)d_out);
}
